// NeuralODE_17944373363329
// MI455X (gfx1250) — matrix operand order repaired and checked by reading, not compiled, not run on hardware
//
#include <hip/hip_runtime.h>

#define UNITS  512
#define MTILE  32          // batch rows per workgroup
#define LDB    520         // bf16 LDS row stride (padded, bank-conflict free)
#define NSTEPS 32

typedef __attribute__((ext_vector_type(16))) __bf16   bf16x16;
typedef __attribute__((ext_vector_type(8)))  float    v8f;
typedef __attribute__((ext_vector_type(4)))  unsigned uint4v;

union Frag { uint4v u[2]; bf16x16 v; };

// Hardware tanh on gfx1250 (V_TANH_F32, single TRANS op) with safe fallback.
#if defined(__has_builtin)
#if __has_builtin(__builtin_amdgcn_tanhf)
#define FAST_TANH(x) __builtin_amdgcn_tanhf(x)
#endif
#endif
#ifndef FAST_TANH
#define FAST_TANH(x) tanhf(x)
#endif

__device__ __forceinline__ unsigned short f2bf(float f) {
    unsigned u = __float_as_uint(f);
    u += 0x7fffu + ((u >> 16) & 1u);   // round-to-nearest-even
    return (unsigned short)(u >> 16);
}

// ---------------------------------------------------------------------------
// Pre-pass: convert fp32 W[512][512] into bf16 packed in the contiguous-k WMMA
// B-fragment layout (16x16x32 bf16).  Fragment (kb, nt):
//   lane l (lo=l&15, hi=l>>4): column n = nt*16+lo,
//   dword dw holds {W[k][n], W[k+1][n]} with k = kb*32 + hi*16 + 2*dw.
// Packed offset: ((kb*32 + nt)*32 + lane)*8 + dw  (dwords).
// ---------------------------------------------------------------------------
__global__ void pack_w_kernel(const float* __restrict__ W, unsigned* __restrict__ out) {
    int p = blockIdx.x * blockDim.x + threadIdx.x;
    if (p >= UNITS * UNITS / 2) return;
    int dw   = p & 7;
    int lane = (p >> 3) & 31;
    int nt   = (p >> 8) & 31;
    int kb   = p >> 13;
    int lo = lane & 15, hi = lane >> 4;
    int n = nt * 16 + lo;
    int k = kb * 32 + hi * 16 + dw * 2;
    unsigned lobits = f2bf(W[k * UNITS + n]);
    unsigned hibits = f2bf(W[(k + 1) * UNITS + n]);
    out[p] = lobits | (hibits << 16);
}

// ---------------------------------------------------------------------------
// One 16x128 output chunk of a [32 x 512] * [512 x 512] GEMM for one wave.
// A from LDS (bf16, row-major, stride LDB), B from packed global weights
// (L2-resident).  Accumulators arrive pre-initialized with the bias.
// ---------------------------------------------------------------------------
__device__ __forceinline__ void gemm_wave(const unsigned short* __restrict__ A,
                                          const unsigned* __restrict__ Bp,
                                          v8f acc[8], int arow, int ntBase,
                                          int lane, int hi) {
#pragma unroll 2
    for (int kb = 0; kb < 16; ++kb) {
        Frag a;
        const unsigned short* ap = A + arow * LDB + kb * 32 + hi * 16;
        a.u[0] = *(const uint4v*)(ap);        // VGPR0-3: K = base .. base+7
        a.u[1] = *(const uint4v*)(ap + 8);    // VGPR4-7: K = base+8 .. base+15
        const unsigned* bbase = Bp + kb * 8192 + ntBase * 256 + lane * 8;
#pragma unroll
        for (int t = 0; t < 8; ++t) {
            Frag b;
            const unsigned* bp = bbase + t * 256;
            b.u[0] = *(const uint4v*)(bp);
            b.u[1] = *(const uint4v*)(bp + 4);
            acc[t] = __builtin_amdgcn_wmma_f32_16x16x32_bf16(
                false, a.v, false, b.v, (short)0, acc[t], false, false);
        }
    }
}

// ---------------------------------------------------------------------------
// Main kernel: each workgroup integrates a 32-row batch tile through all
// 32 RK4 steps.  bf16 activations live in LDS; fp32 RK4 state (y, k-accum)
// lives in the owning thread's registers (ownership == WMMA C layout).
// ---------------------------------------------------------------------------
__global__ __launch_bounds__(256)
void node_rk4_kernel(const float* __restrict__ x,
                     const float* __restrict__ times,
                     const float* __restrict__ b1,
                     const float* __restrict__ b2,
                     const unsigned* __restrict__ W1p,
                     const unsigned* __restrict__ W2p,
                     float* __restrict__ out) {
    extern __shared__ char smem[];
    unsigned short* In  = (unsigned short*)smem;              // 32 x 520 bf16
    unsigned short* Hid = In + MTILE * LDB;                   // 32 x 520 bf16

    const int tid    = threadIdx.x;
    const int lane   = tid & 31;
    const int wave   = tid >> 5;
    const int lo     = lane & 15;
    const int hi     = lane >> 4;
    const int mstrip = wave & 1;            // which 16-row strip
    const int ntBase = (wave >> 1) * 8;     // first 16-col tile of this wave
    const int arow   = mstrip * 16 + lo;
    const int row0   = blockIdx.x * MTILE;

    const float dt = (times[1] - times[0]) / (float)NSTEPS;
    const float wc[4] = { dt / 6.0f, dt / 3.0f, dt / 3.0f, dt / 6.0f };
    const float cc[3] = { 0.5f * dt, 0.5f * dt, dt };

    // Per-lane bias (one value per owned 16x16 tile; same for all 8 rows).
    float b1r[8], b2r[8];
#pragma unroll
    for (int t = 0; t < 8; ++t) {
        int n = (ntBase + t) * 16 + lo;
        b1r[t] = b1[n];
        b2r[t] = b2[n];
    }

    // Per-thread fp32 state: element (t,r) <-> (m, n) of the WMMA C layout.
    float ys[8][8];    // master ODE state y
    float accs[8][8];  // RK4 weighted k accumulator
#pragma unroll
    for (int t = 0; t < 8; ++t) {
        int n = (ntBase + t) * 16 + lo;
#pragma unroll
        for (int r = 0; r < 8; ++r) {
            int m = mstrip * 16 + hi * 8 + r;
            float v = x[(row0 + m) * UNITS + n];
            ys[t][r] = v;
            In[m * LDB + n] = f2bf(v);
        }
    }
    __syncthreads();

    for (int step = 0; step < NSTEPS; ++step) {
#pragma unroll 1
        for (int s = 0; s < 4; ++s) {
            // ---- layer 1: Hid = bf16(tanh(In @ W1 + b1)) ----
            v8f acc[8];
#pragma unroll
            for (int t = 0; t < 8; ++t) acc[t] = (v8f)(b1r[t]);
            gemm_wave(In, W1p, acc, arow, ntBase, lane, hi);
#pragma unroll
            for (int t = 0; t < 8; ++t) {
                int n = (ntBase + t) * 16 + lo;
#pragma unroll
                for (int r = 0; r < 8; ++r) {
                    int m = mstrip * 16 + hi * 8 + r;
                    Hid[m * LDB + n] = f2bf(FAST_TANH(acc[t][r]));
                }
            }
            __syncthreads();

            // ---- layer 2: k = tanh(Hid @ W2 + b2); RK4 stage update ----
#pragma unroll
            for (int t = 0; t < 8; ++t) acc[t] = (v8f)(b2r[t]);
            gemm_wave(Hid, W2p, acc, arow, ntBase, lane, hi);
#pragma unroll
            for (int t = 0; t < 8; ++t) {
                int n = (ntBase + t) * 16 + lo;
#pragma unroll
                for (int r = 0; r < 8; ++r) {
                    int m = mstrip * 16 + hi * 8 + r;
                    float k = FAST_TANH(acc[t][r]);
                    if (s == 0) accs[t][r] = wc[0] * k;
                    else        accs[t][r] = fmaf(wc[s], k, accs[t][r]);
                    if (s < 3) {
                        In[m * LDB + n] = f2bf(fmaf(cc[s], k, ys[t][r]));
                    } else {
                        float yn = ys[t][r] + accs[t][r];
                        ys[t][r] = yn;
                        In[m * LDB + n] = f2bf(yn);
                    }
                }
            }
            __syncthreads();
        }
    }

    // Store final state from registers.
#pragma unroll
    for (int t = 0; t < 8; ++t) {
        int n = (ntBase + t) * 16 + lo;
#pragma unroll
        for (int r = 0; r < 8; ++r) {
            int m = mstrip * 16 + hi * 8 + r;
            out[(row0 + m) * UNITS + n] = ys[t][r];
        }
    }
}

// ---------------------------------------------------------------------------
extern "C" void kernel_launch(void* const* d_in, const int* in_sizes, int n_in,
                              void* d_out, int out_size, void* d_ws, size_t ws_size,
                              hipStream_t stream) {
    const float* x     = (const float*)d_in[0];
    const float* times = (const float*)d_in[1];
    const float* W1    = (const float*)d_in[2];
    const float* b1    = (const float*)d_in[3];
    const float* W2    = (const float*)d_in[4];
    const float* b2    = (const float*)d_in[5];
    float* out = (float*)d_out;

    unsigned* W1p = (unsigned*)d_ws;                       // 512 KB packed bf16
    unsigned* W2p = W1p + (UNITS * UNITS / 2);             // 512 KB packed bf16

    const int packN = UNITS * UNITS / 2;                   // dwords per weight
    pack_w_kernel<<<(packN + 255) / 256, 256, 0, stream>>>(W1, W1p);
    pack_w_kernel<<<(packN + 255) / 256, 256, 0, stream>>>(W2, W2p);

    const int batch = in_sizes[0] / UNITS;                 // 2048
    const size_t smem = (size_t)(2 * MTILE * LDB * sizeof(unsigned short));
    (void)hipFuncSetAttribute((const void*)node_rk4_kernel,
                              hipFuncAttributeMaxDynamicSharedMemorySize,
                              (int)smem);
    node_rk4_kernel<<<batch / MTILE, 256, smem, stream>>>(
        x, times, b1, b2, W1p, W2p, out);
}
